// LocalKNN_75711683494137
// MI455X (gfx1250) — matrix operand order repaired and checked by reading, not compiled, not run on hardware
//
#include <hip/hip_runtime.h>

typedef __attribute__((ext_vector_type(16))) __bf16 v16bf;
typedef __attribute__((ext_vector_type(8)))  __bf16 v8bf;
typedef __attribute__((ext_vector_type(8)))  float  v8f;

#define C_DIM   256
#define M_KEYS  65536
#define N_Q     12544
#define HWSZ    3136          // 56*56
#define TOPK    5

#define QT      64            // queries per workgroup
#define KT      32            // keys per LDS tile
#define NT      (M_KEYS / KT) // 2048 tiles
#define NBUF    3             // triple-buffered key tiles
#define LDS_STRIDE 264        // padded bf16 row stride (132 dwords == 4 mod 64 banks)

// LDS layout (byte offsets from dynamic-shared base; kernel has no static LDS,
// so these equal the hardware LDS addresses used by the async-LDS instructions)
#define Q_OFF        0
#define K_OFF        (QT * LDS_STRIDE * 2)              // 33792
#define K_BUF_BYTES  (KT * LDS_STRIDE * 2)              // 16896
#define MERGE_OFF    (K_OFF + NBUF * K_BUF_BYTES)       // 84480
#define SMEM_BYTES   (MERGE_OFF + QT * 32 * TOPK * 4)   // 125440

// Workspace layout (bytes)
#define WS_KEYSB 0u
#define WS_QB    33554432u     // 65536*256*2
#define WS_K2    39976960u     // + 12544*256*2
#define WS_Q2    40239104u     // + 65536*4
#define WS_TMP   40289280u     // + 12544*4

union Frag { v16bf v; v8bf h[2]; };

// 64B global -> LDS async copy (4 x b128). ISA: the same INST_OFFSET is added
// to both the LDS address (%0) and the global address (%1), and our staging
// layout advances both in lockstep, so one address pair covers all four.
__device__ __forceinline__ void async_copy64(unsigned lds_off, const void* gptr) {
  asm volatile("global_load_async_to_lds_b128 %0, %1, off\n\t"
               "global_load_async_to_lds_b128 %0, %1, off offset:16\n\t"
               "global_load_async_to_lds_b128 %0, %1, off offset:32\n\t"
               "global_load_async_to_lds_b128 %0, %1, off offset:48"
               :: "v"(lds_off), "v"(gptr) : "memory");
}
__device__ __forceinline__ void wait_async0() {
  asm volatile("s_wait_asynccnt 0x0" ::: "memory");
}
__device__ __forceinline__ void wait_async4() {
  asm volatile("s_wait_asynccnt 0x4" ::: "memory");
}

// -------------------- prep: keys -> bf16 + ||k||^2 --------------------
extern "C" __global__ void knn_prep_k(const float* __restrict__ keys,
                                      __bf16* __restrict__ kb,
                                      float* __restrict__ k2) {
  __shared__ float red[256];
  const int m = blockIdx.x;
  const int c = threadIdx.x;
  float x = keys[(size_t)m * C_DIM + c];
  kb[(size_t)m * C_DIM + c] = (__bf16)x;
  red[c] = x * x;
  __syncthreads();
  for (int s = 128; s > 0; s >>= 1) {
    if (c < s) red[c] += red[c + s];
    __syncthreads();
  }
  if (c == 0) k2[m] = red[0];
}

// ------------- prep: feat_map [B,C,H,W] -> q bf16 [N,C] + ||q||^2 -------------
extern "C" __global__ void knn_prep_q(const float* __restrict__ feat,
                                      __bf16* __restrict__ qb,
                                      float* __restrict__ q2) {
  __shared__ float red[256];
  const int n  = blockIdx.x;       // query index, n = (b*H + h)*W + w
  const int c  = threadIdx.x;
  const int b  = n / HWSZ;
  const int hw = n % HWSZ;
  float x = feat[(size_t)(b * C_DIM + c) * HWSZ + hw];
  qb[(size_t)n * C_DIM + c] = (__bf16)x;
  red[c] = x * x;
  __syncthreads();
  for (int s = 128; s > 0; s >>= 1) {
    if (c < s) red[c] += red[c + s];
    __syncthreads();
  }
  if (c == 0) q2[n] = red[0];
}

// -------------------- main: WMMA distance GEMM + streaming top-5 --------------------
extern "C" __global__ void __launch_bounds__(256, 1)
knn_main(const __bf16* __restrict__ keysb,
         const __bf16* __restrict__ qb,
         const float* __restrict__ k2g,
         const float* __restrict__ q2g,
         float* __restrict__ tmp) {
  extern __shared__ char smem[];
  __bf16* qlds  = (__bf16*)(smem + Q_OFF);
  float*  merge = (float*)(smem + MERGE_OFF);

  const int tid     = threadIdx.x;
  const int lane    = tid & 31;
  const int wave    = tid >> 5;       // 8 waves
  const int lane_lo = lane & 15;
  const int half    = lane >> 4;
  const int rowblk  = wave >> 1;      // 0..3 : which 16 queries
  const int colblk  = wave & 1;       // 0..1 : which 16 keys of the 32-key tile
  const int qbase   = blockIdx.x * QT;

  // ---- stage queries into LDS: 64 rows x 256 bf16 ----
  {
    const int qr = tid >> 2;          // 0..63
    const int c0 = (tid & 3) * 64;
    const __bf16* src = qb + (size_t)(qbase + qr) * C_DIM + c0;
    __bf16* dst = qlds + qr * LDS_STRIDE + c0;
#pragma unroll
    for (int i = 0; i < 8; ++i)
      *(v8bf*)(dst + i * 8) = *(const v8bf*)(src + i * 8);
  }

  // ---- stage key tiles 0 and 1 (async global -> LDS, triple-buffered) ----
  const int kr  = tid >> 3;           // 0..31 key row in tile
  const int kc0 = (tid & 7) * 32;     // 32 channels per thread
  const unsigned lds_stage0 = (unsigned)(K_OFF + kr * LDS_STRIDE * 2 + kc0 * 2);
  async_copy64(lds_stage0, keysb + (size_t)kr * C_DIM + kc0);
  async_copy64(lds_stage0 + K_BUF_BYTES,
               keysb + ((size_t)KT + kr) * C_DIM + kc0);
  wait_async4();                      // tile 0 complete (tile 1 may be in flight)
  __syncthreads();

  // ---- preload A fragments (contiguous K order matching B: lanes<16 K 0..15, lanes>=16 K 16..31) ----
  Frag a[8];
  {
    const __bf16* arow = qlds + (rowblk * 16 + lane_lo) * LDS_STRIDE;
#pragma unroll
    for (int kk = 0; kk < 8; ++kk) {
      a[kk].h[0] = *(const v8bf*)(arow + kk * 32 + half * 16);
      a[kk].h[1] = *(const v8bf*)(arow + kk * 32 + half * 16 + 8);
    }
  }

  float best[8][TOPK];
#pragma unroll
  for (int r = 0; r < 8; ++r)
#pragma unroll
    for (int j = 0; j < TOPK; ++j) best[r][j] = __builtin_inff();

  // LDS byte address of this wave's B row (before buffer offset)
  const unsigned brow_row = (unsigned)(K_OFF +
                    (colblk * 16 + lane_lo) * LDS_STRIDE * 2 + half * 32);

  int cur = 0;                        // t % NBUF
  for (int t = 0; t < NT; ++t) {
    // issue tile t+2 into buffer (cur+2)%3 (overlaps with WMMA below)
    const bool issue = (t + 2) < NT;
    if (issue) {
      int sbuf = cur + 2; if (sbuf >= NBUF) sbuf -= NBUF;
      const __bf16* src = keysb + ((size_t)(t + 2) * KT + kr) * C_DIM + kc0;
      async_copy64(lds_stage0 + (unsigned)(sbuf * K_BUF_BYTES), src);
      if (t + 12 < NT)
        __builtin_prefetch(keysb + ((size_t)(t + 12) * KT + kr) * C_DIM + kc0, 0, 1);
    }

    // ---- 16x16x256 tile: clause-load all 8 B frags (one wait), then 8 chained WMMAs ----
    // B 32x16 layout: lanes<16 hold K 0..15 contiguous, lanes>=16 hold K 16..31.
    // chunk kk at bytes kk*64 (+16); +32B gap is the other half-wave's K range.
    Frag b[8];
    {
      const unsigned ba = brow_row + (unsigned)(cur * K_BUF_BYTES);
      asm volatile(
        "s_clause 0xf\n\t"
        "ds_load_b128 %[x0], %[a] offset:0\n\t"
        "ds_load_b128 %[x1], %[a] offset:16\n\t"
        "ds_load_b128 %[x2], %[a] offset:64\n\t"
        "ds_load_b128 %[x3], %[a] offset:80\n\t"
        "ds_load_b128 %[x4], %[a] offset:128\n\t"
        "ds_load_b128 %[x5], %[a] offset:144\n\t"
        "ds_load_b128 %[x6], %[a] offset:192\n\t"
        "ds_load_b128 %[x7], %[a] offset:208\n\t"
        "ds_load_b128 %[x8], %[a] offset:256\n\t"
        "ds_load_b128 %[x9], %[a] offset:272\n\t"
        "ds_load_b128 %[xa], %[a] offset:320\n\t"
        "ds_load_b128 %[xb], %[a] offset:336\n\t"
        "ds_load_b128 %[xc], %[a] offset:384\n\t"
        "ds_load_b128 %[xd], %[a] offset:400\n\t"
        "ds_load_b128 %[xe], %[a] offset:448\n\t"
        "ds_load_b128 %[xf], %[a] offset:464\n\t"
        "s_wait_dscnt 0x0"
        : [x0]"=v"(b[0].h[0]), [x1]"=v"(b[0].h[1]),
          [x2]"=v"(b[1].h[0]), [x3]"=v"(b[1].h[1]),
          [x4]"=v"(b[2].h[0]), [x5]"=v"(b[2].h[1]),
          [x6]"=v"(b[3].h[0]), [x7]"=v"(b[3].h[1]),
          [x8]"=v"(b[4].h[0]), [x9]"=v"(b[4].h[1]),
          [xa]"=v"(b[5].h[0]), [xb]"=v"(b[5].h[1]),
          [xc]"=v"(b[6].h[0]), [xd]"=v"(b[6].h[1]),
          [xe]"=v"(b[7].h[0]), [xf]"=v"(b[7].h[1])
        : [a]"v"(ba)
        : "memory");
    }
    v8f acc = {};
#pragma unroll
    for (int kk = 0; kk < 8; ++kk)
      acc = __builtin_amdgcn_wmma_f32_16x16x32_bf16(false, a[kk].v, false, b[kk].v,
                                                    (short)0, acc, false, false);

    // rank by s = ||k||^2 - 2*q.k  (||q||^2 added at the end)
    const float k2v = k2g[t * KT + colblk * 16 + lane_lo];
#pragma unroll
    for (int r = 0; r < 8; ++r) {
      float d = k2v - 2.0f * acc[r];
      if (d < best[r][TOPK - 1]) {       // rare -> cheap guard
        float v = d;
#pragma unroll
        for (int j = 0; j < TOPK; ++j) {
          float lo = fminf(best[r][j], v);
          float hi = fmaxf(best[r][j], v);
          best[r][j] = lo;
          v = hi;
        }
      }
    }

    // ensure tile t+1 landed in LDS (t+2 may stay in flight), then rotate
    if (issue) wait_async4();
    else       wait_async0();
    __syncthreads();
    cur = cur + 1; if (cur >= NBUF) cur -= NBUF;
  }

  // ---- merge lane-local top-5s: [64 rows][32 slots][5] ----
  {
    const int slot = colblk * 16 + lane_lo;
#pragma unroll
    for (int r = 0; r < 8; ++r) {
      const int row_local = rowblk * 16 + 8 * half + r;   // C/D layout: VGPR r -> M = r + 8*half
#pragma unroll
      for (int j = 0; j < TOPK; ++j)
        merge[(row_local * 32 + slot) * TOPK + j] = best[r][j];
    }
  }
  __syncthreads();

  if (tid < QT) {
    float b5[TOPK];
#pragma unroll
    for (int j = 0; j < TOPK; ++j) b5[j] = __builtin_inff();
    const float* row = merge + tid * (32 * TOPK);
    for (int s = 0; s < 32 * TOPK; ++s) {
      float v = row[s];
      if (v < b5[TOPK - 1]) {
        float x = v;
#pragma unroll
        for (int j = 0; j < TOPK; ++j) {
          float lo = fminf(b5[j], x);
          float hi = fmaxf(b5[j], x);
          b5[j] = lo;
          x = hi;
        }
      }
    }
    float mean = (b5[0] + b5[1] + b5[2] + b5[3] + b5[4]) * 0.2f + q2g[qbase + tid];
    tmp[qbase + tid] = mean;
  }
}

// -------------------- per-image min/max normalize --------------------
extern "C" __global__ void knn_norm(const float* __restrict__ tmp,
                                    float* __restrict__ out) {
  __shared__ float smin[256];
  __shared__ float smax[256];
  const int b   = blockIdx.x;
  const int tid = threadIdx.x;
  float mn = __builtin_inff(), mx = -__builtin_inff();
  for (int i = tid; i < HWSZ; i += 256) {
    float v = tmp[b * HWSZ + i];
    mn = fminf(mn, v);
    mx = fmaxf(mx, v);
  }
  smin[tid] = mn; smax[tid] = mx;
  __syncthreads();
  for (int s = 128; s > 0; s >>= 1) {
    if (tid < s) {
      smin[tid] = fminf(smin[tid], smin[tid + s]);
      smax[tid] = fmaxf(smax[tid], smax[tid + s]);
    }
    __syncthreads();
  }
  const float vmin = smin[0], vmax = smax[0];
  const float inv = 1.0f / (vmax - vmin + 1e-6f);
  for (int i = tid; i < HWSZ; i += 256)
    out[b * HWSZ + i] = (tmp[b * HWSZ + i] - vmin) * inv;
}

// -------------------- launch --------------------
extern "C" void kernel_launch(void* const* d_in, const int* in_sizes, int n_in,
                              void* d_out, int out_size, void* d_ws, size_t ws_size,
                              hipStream_t stream) {
  const float* feat = (const float*)d_in[0];   // [4,256,56,56] f32
  const float* keys = (const float*)d_in[1];   // [65536,256]   f32
  float* out = (float*)d_out;                  // [4,56,56]     f32
  char* ws = (char*)d_ws;

  __bf16* keysb = (__bf16*)(ws + WS_KEYSB);
  __bf16* qb    = (__bf16*)(ws + WS_QB);
  float*  k2    = (float*)(ws + WS_K2);
  float*  q2    = (float*)(ws + WS_Q2);
  float*  tmp   = (float*)(ws + WS_TMP);

  knn_prep_k<<<M_KEYS, 256, 0, stream>>>(keys, keysb, k2);
  knn_prep_q<<<N_Q, 256, 0, stream>>>(feat, qb, q2);
  knn_main<<<N_Q / QT, 256, SMEM_BYTES, stream>>>(keysb, qb, k2, q2, tmp);
  knn_norm<<<4, 256, 0, stream>>>(tmp, out);
}
